// GenericAttention_42460046688952
// MI455X (gfx1250) — matrix operand order repaired and checked by reading, not compiled, not run on hardware
//
#include <hip/hip_runtime.h>
#include <hip/hip_bf16.h>

typedef __attribute__((ext_vector_type(16))) _Float16 v16h;
typedef __attribute__((ext_vector_type(8)))  _Float16 v8h;
typedef __attribute__((ext_vector_type(4)))  _Float16 v4h;
typedef __attribute__((ext_vector_type(8)))  float    v8f;
typedef __attribute__((ext_vector_type(4)))  int      v4i;
typedef __attribute__((address_space(1)))    v4i      g_v4i;  // global int4
typedef __attribute__((address_space(3)))    v4i      l_v4i;  // LDS int4

#if defined(__gfx1250__) && __has_builtin(__builtin_amdgcn_global_load_async_to_lds_b128)
#define USE_ASYNC_LDS 1
#else
#define USE_ASYNC_LDS 0
#endif

constexpr int BATCH = 2, SEQ = 2048, EMB = 2048, NH = 16, NKVH = 8, HD = 128;
constexpr int GQ = NH / NKVH;          // 2
constexpr int MROWS = BATCH * SEQ;     // 4096
constexpr int N3 = 3 * NH * HD;        // 6144
constexpr float MASK_VALUE = -0.7f * 3.402823466e38f;

// ---- CDNA5 wave32 WMMA fragment helpers (16x16x32 f16, f32 accum) ----
// A (16Mx32K): lane m=lane&15, half=lane>>4; VGPR i holds K pair at
// kb = (i/4)*16 + (i%4)*2 + half*8.
__device__ __forceinline__ v16h load_a_frag(const _Float16* a, int lda) {
  int lane = threadIdx.x & 31;
  int mr = lane & 15, half = lane >> 4;
  const _Float16* row = a + mr * lda + half * 8;
  v16h f;
#pragma unroll
  for (int i = 0; i < 8; ++i) {
    int kb = (i >> 2) * 16 + (i & 3) * 2;
    f[2 * i]     = row[kb];
    f[2 * i + 1] = row[kb + 1];
  }
  return f;
}

// B (32Kx16N), source n-major (bt[n*ldb + k]): lane n=lane&15, half=lane>>4,
// element j holds K = half*8 + (j/8)*16 + j%8 (two 8-half runs -> ds_load_b128 pair).
__device__ __forceinline__ v16h load_b_frag(const _Float16* bt, int ldb) {
  int lane = threadIdx.x & 31;
  int n = lane & 15, half = lane >> 4;
  const _Float16* col = bt + n * ldb + half * 8;
  v16h f;
#pragma unroll
  for (int i = 0; i < 16; ++i) f[i] = col[(i >> 3) * 16 + (i & 7)];
  return f;
}

__device__ __forceinline__ v8f wmma_f16(v16h a, v16h b, v8f c) {
  return __builtin_amdgcn_wmma_f32_16x16x32_f16(false, a, false, b, (short)0, c,
                                                false, false);
}

// ============================================================================
// Kernel 1: QKV projection GEMM (f16 WMMA) + register-resident RoPE.
// Wave-group 0 owns column subtiles {0,1,4,5}, group 1 owns {2,3,6,7}, so each
// lane holds both halves of every RoPE pair (acc[j], acc[j+2]) in registers.
// Grid: (MROWS/64, 32 live (c,h) pairs). Dead KV heads (h>=8) never launched.
// ============================================================================
__global__ __launch_bounds__(256) void qkv_rope_kernel(
    const float* __restrict__ xq,    // (MROWS, EMB)
    const int* __restrict__ xpos,    // (BATCH, SEQ)
    const float* __restrict__ wqkv,  // (EMB, N3)
    _Float16* __restrict__ qout,     // (BATCH, NH,   SEQ, HD)
    _Float16* __restrict__ kout,     // (BATCH, NKVH, SEQ, HD)
    _Float16* __restrict__ vout) {   // (BATCH, NKVH, SEQ, HD)
  __shared__ __align__(16) _Float16 As[64 * 40];   // [m][k] lda=40
  __shared__ __align__(16) _Float16 Bs[128 * 40];  // [n][k] ldb=40 (transposed)

  const int tid = threadIdx.x;
  const int wave = tid >> 5, lane = tid & 31;
  const int mr = lane & 15, half = lane >> 4;
  const int wm = wave & 3, grp = wave >> 2;
  int jcols[4];
#pragma unroll
  for (int i = 0; i < 4; ++i) jcols[i] = (i & 1) + (i >> 1) * 4 + grp * 2;

  const int ch = blockIdx.y;
  int c, h;
  if (ch < 16)      { c = 0; h = ch; }
  else if (ch < 24) { c = 1; h = ch - 16; }
  else              { c = 2; h = ch - 24; }
  const int n0 = c * NH * HD + h * HD;
  const int m0 = blockIdx.x * 64;

  float4 aR[2], bR[4];
  auto fetch = [&](int kk) {
#pragma unroll
    for (int i = 0; i < 2; ++i) {
      int s = tid + i * 256, r = s >> 3, c4 = s & 7;
      aR[i] = *(const float4*)(xq + (size_t)(m0 + r) * EMB + kk + c4 * 4);
    }
#pragma unroll
    for (int i = 0; i < 4; ++i) {
      int s = tid + i * 256, kr = s >> 5, nc4 = s & 31;
      bR[i] = *(const float4*)(wqkv + (size_t)(kk + kr) * N3 + n0 + nc4 * 4);
    }
  };
  auto stage = [&]() {
#pragma unroll
    for (int i = 0; i < 2; ++i) {
      int s = tid + i * 256, r = s >> 3, c4 = s & 7;
      v4h h4 = {(_Float16)aR[i].x, (_Float16)aR[i].y, (_Float16)aR[i].z,
                (_Float16)aR[i].w};
      *(v4h*)(As + r * 40 + c4 * 4) = h4;
    }
#pragma unroll
    for (int i = 0; i < 4; ++i) {
      int s = tid + i * 256, kr = s >> 5, nc4 = s & 31;
      Bs[(nc4 * 4 + 0) * 40 + kr] = (_Float16)bR[i].x;
      Bs[(nc4 * 4 + 1) * 40 + kr] = (_Float16)bR[i].y;
      Bs[(nc4 * 4 + 2) * 40 + kr] = (_Float16)bR[i].z;
      Bs[(nc4 * 4 + 3) * 40 + kr] = (_Float16)bR[i].w;
    }
  };

  v8f acc[4] = {};
  fetch(0);
  for (int kk = 0; kk < EMB; kk += 32) {
    stage();
    __syncthreads();
    if (kk + 32 < EMB) fetch(kk + 32);  // overlap next global fetch with WMMA
    v16h af = load_a_frag(&As[wm * 16 * 40], 40);
#pragma unroll
    for (int j = 0; j < 4; ++j) {
      v16h bf = load_b_frag(&Bs[jcols[j] * 16 * 40], 40);
      acc[j] = wmma_f16(af, bf, acc[j]);
    }
    __syncthreads();
  }

  const int b = m0 / SEQ;  // 64-row tile never crosses batch boundary
  const int t0 = (m0 % SEQ) + wm * 16 + 8 * half;
  if (c < 2) {
    _Float16* outp = (c == 0) ? qout : kout;
    const int nheads = (c == 0) ? NH : NKVH;
#pragma unroll
    for (int ji = 0; ji < 2; ++ji) {
      int d = jcols[ji] * 16 + mr;  // in [0,64); pair partner is d+64
      float inv_ts = __powf(10000.f, (float)d * (-1.0f / 64.0f));
#pragma unroll
      for (int r = 0; r < 8; ++r) {
        int t = t0 + r;
        float p = (float)xpos[b * SEQ + t];
        float sn, cs;
        __sincosf(p * inv_ts, &sn, &cs);
        float v1 = acc[ji][r], v2 = acc[ji + 2][r];
        size_t base = ((size_t)(b * nheads + h) * SEQ + t) * HD;
        outp[base + d]      = (_Float16)(v1 * cs - v2 * sn);
        outp[base + d + 64] = (_Float16)(v2 * cs + v1 * sn);
      }
    }
  } else {
#pragma unroll
    for (int j = 0; j < 4; ++j) {
      int d = jcols[j] * 16 + mr;
#pragma unroll
      for (int r = 0; r < 8; ++r) {
        int t = t0 + r;
        vout[((size_t)(b * NKVH + h) * SEQ + t) * HD + d] = (_Float16)acc[j][r];
      }
    }
  }
}

// ============================================================================
// Kernel 2: causal GQA flash attention, f16 WMMA, online softmax.
// K tile staged with gfx1250 async global->LDS DMA (ASYNCcnt) when available.
// Grid: (SEQ/128, BATCH*NH). 8 waves; wave w owns q rows [w*16, w*16+16).
// ============================================================================
__global__ __launch_bounds__(256) void attn_kernel(
    const _Float16* __restrict__ q,  // (BATCH, NH,   SEQ, HD)
    const _Float16* __restrict__ k,  // (BATCH, NKVH, SEQ, HD)
    const _Float16* __restrict__ v,  // (BATCH, NKVH, SEQ, HD)
    const int* __restrict__ seg,     // (BATCH, SEQ)
    _Float16* __restrict__ o) {      // (MROWS, NH*HD)
  __shared__ __align__(16) _Float16 Ks[64 * 136];    // [kv][d]  ld=136 (16B rows)
  __shared__ __align__(16) _Float16 Vt[128 * 72];    // [d][kv]  ld=72
  __shared__ __align__(16) _Float16 Ps[8 * 16 * 72]; // per-wave P [m][kv]
  __shared__ int seg_kv_s[64];

  const int tid = threadIdx.x;
  const int wave = tid >> 5, lane = tid & 31;
  const int mr = lane & 15, half = lane >> 4;

  const int bh = blockIdx.y;
  const int b = bh / NH, h = bh % NH;
  const int kvh = h / GQ;
  const int q0 = blockIdx.x * 128;
  const int qrow_base = q0 + wave * 16;

  const _Float16* qptr = q + ((size_t)(b * NH + h) * SEQ + qrow_base) * HD;
  v16h qf[4];
#pragma unroll
  for (int cc = 0; cc < 4; ++cc) qf[cc] = load_a_frag(qptr + cc * 32, HD);

  int seg_q[8];
#pragma unroll
  for (int r = 0; r < 8; ++r)
    seg_q[r] = seg[b * SEQ + qrow_base + r + 8 * half];

  v8f oacc[8] = {};
  float mrow[8], lrow[8];
#pragma unroll
  for (int r = 0; r < 8; ++r) { mrow[r] = -INFINITY; lrow[r] = 0.0f; }

  const _Float16* kbase = k + (size_t)(b * NKVH + kvh) * SEQ * HD;
  const _Float16* vbase = v + (size_t)(b * NKVH + kvh) * SEQ * HD;

  for (int kv0 = 0; kv0 < q0 + 128; kv0 += 64) {
    // --- K tile: pure byte copy -> async DMA global->LDS when available ---
#if USE_ASYNC_LDS
#pragma unroll
    for (int i = 0; i < 4; ++i) {
      int s = tid + i * 256, r = s >> 4, c8 = s & 15;
      __builtin_amdgcn_global_load_async_to_lds_b128(
          (g_v4i*)(kbase + (size_t)(kv0 + r) * HD + c8 * 8),
          (l_v4i*)(&Ks[r * 136 + c8 * 8]), 0, 0);
    }
#else
#pragma unroll
    for (int i = 0; i < 4; ++i) {
      int s = tid + i * 256, r = s >> 4, c8 = s & 15;
      *(v8h*)(&Ks[r * 136 + c8 * 8]) =
          *(const v8h*)(kbase + (size_t)(kv0 + r) * HD + c8 * 8);
    }
#endif
    // --- V tile: b128 load, transpose at f16 granularity while staging ---
#pragma unroll
    for (int i = 0; i < 4; ++i) {
      int s = tid + i * 256, r = s >> 4, c8 = s & 15;
      v8h vv = *(const v8h*)(vbase + (size_t)(kv0 + r) * HD + c8 * 8);
#pragma unroll
      for (int j = 0; j < 8; ++j) Vt[(c8 * 8 + j) * 72 + r] = vv[j];
    }
    if (tid < 64) seg_kv_s[tid] = seg[b * SEQ + kv0 + tid];
    if (kv0 + 64 < q0 + 128) {  // prefetch next KV tile (global_prefetch_b8)
      __builtin_prefetch(kbase + (size_t)(kv0 + 64) * HD + tid * 32, 0, 0);
      __builtin_prefetch(vbase + (size_t)(kv0 + 64) * HD + tid * 32, 0, 0);
    }
#if USE_ASYNC_LDS
#if __has_builtin(__builtin_amdgcn_s_wait_asynccnt)
    __builtin_amdgcn_s_wait_asynccnt(0);
#else
    asm volatile("s_wait_asynccnt 0x0" ::: "memory");
#endif
#endif
    __syncthreads();

    if (kv0 <= qrow_base + 15) {  // wave-uniform causal skip (EXEC stays full)
      v8f sacc[4];
#pragma unroll
      for (int j = 0; j < 4; ++j) {
        v8f sa = {0.f, 0.f, 0.f, 0.f, 0.f, 0.f, 0.f, 0.f};
#pragma unroll
        for (int cc = 0; cc < 4; ++cc) {
          v16h bf = load_b_frag(&Ks[(j * 16) * 136 + cc * 32], 136);
          sa = wmma_f16(qf[cc], bf, sa);
        }
        sacc[j] = sa;
      }

      float rmax[8];
#pragma unroll
      for (int r = 0; r < 8; ++r) {
        int trow = qrow_base + r + 8 * half;
        float mx = -INFINITY;
#pragma unroll
        for (int j = 0; j < 4; ++j) {
          int scol = kv0 + j * 16 + mr;
          bool ok = (scol <= trow) && (seg_kv_s[j * 16 + mr] == seg_q[r]);
          float val = ok ? sacc[j][r] : MASK_VALUE;
          sacc[j][r] = val;
          mx = fmaxf(mx, val);
        }
#pragma unroll
        for (int off = 1; off < 16; off <<= 1)
          mx = fmaxf(mx, __shfl_xor(mx, off, 32));
        rmax[r] = mx;
      }

#pragma unroll
      for (int r = 0; r < 8; ++r) {
        float mnew = fmaxf(mrow[r], rmax[r]);
        float corr = __expf(mrow[r] - mnew);
        mrow[r] = mnew;
        float rs = 0.0f;
#pragma unroll
        for (int j = 0; j < 4; ++j) {
          float e = __expf(sacc[j][r] - mnew);
          sacc[j][r] = e;
          rs += e;
        }
#pragma unroll
        for (int off = 1; off < 16; off <<= 1) rs += __shfl_xor(rs, off, 32);
        lrow[r] = lrow[r] * corr + rs;
#pragma unroll
        for (int jn = 0; jn < 8; ++jn) oacc[jn][r] *= corr;
      }

      // Re-lay P: C-fragment -> A-fragment via per-wave LDS scratch.
      _Float16* pw = &Ps[wave * 16 * 72];
#pragma unroll
      for (int j = 0; j < 4; ++j)
#pragma unroll
        for (int r = 0; r < 8; ++r)
          pw[(r + 8 * half) * 72 + j * 16 + mr] = (_Float16)sacc[j][r];

#pragma unroll
      for (int cc = 0; cc < 2; ++cc) {
        v16h pf = load_a_frag(pw + cc * 32, 72);
#pragma unroll
        for (int jn = 0; jn < 8; ++jn) {
          v16h bf = load_b_frag(&Vt[(jn * 16) * 72 + cc * 32], 72);
          oacc[jn] = wmma_f16(pf, bf, oacc[jn]);
        }
      }
    }
    __syncthreads();
  }

#pragma unroll
  for (int r = 0; r < 8; ++r) {
    float inv = 1.0f / lrow[r];
    int trow = qrow_base + r + 8 * half;
    size_t base = (size_t)(b * SEQ + trow) * (NH * HD) + (size_t)h * HD;
#pragma unroll
    for (int jn = 0; jn < 8; ++jn)
      o[base + jn * 16 + mr] = (_Float16)(oacc[jn][r] * inv);
  }
}

// ============================================================================
// Kernel 3: output projection: (MROWS x 2048) f16 x (2048 x EMB) f32 -> f32.
// Software-pipelined like kernel 1. Grid: (MROWS/64, EMB/128).
// ============================================================================
__global__ __launch_bounds__(256) void outproj_kernel(
    const _Float16* __restrict__ attn,  // (MROWS, NH*HD)
    const float* __restrict__ wout,     // (NH*HD, EMB)
    float* __restrict__ out) {          // (MROWS, EMB)
  __shared__ __align__(16) _Float16 As[64 * 40];
  __shared__ __align__(16) _Float16 Bs[128 * 40];

  const int tid = threadIdx.x;
  const int wave = tid >> 5, lane = tid & 31;
  const int mr = lane & 15, half = lane >> 4;
  const int wm = wave & 3, wn = (wave >> 2) * 4;
  const int m0 = blockIdx.x * 64;
  const int n0 = blockIdx.y * 128;
  constexpr int KD = NH * HD;  // 2048

  const int ar = tid >> 2, ac8 = tid & 3;  // A: one v8h per thread
  v8h aR;
  float4 bR[4];
  auto fetch = [&](int kk) {
    aR = *(const v8h*)(attn + (size_t)(m0 + ar) * KD + kk + ac8 * 8);
#pragma unroll
    for (int i = 0; i < 4; ++i) {
      int s = tid + i * 256, kr = s >> 5, nc4 = s & 31;
      bR[i] = *(const float4*)(wout + (size_t)(kk + kr) * EMB + n0 + nc4 * 4);
    }
  };
  auto stage = [&]() {
    *(v8h*)(As + ar * 40 + ac8 * 8) = aR;
#pragma unroll
    for (int i = 0; i < 4; ++i) {
      int s = tid + i * 256, kr = s >> 5, nc4 = s & 31;
      Bs[(nc4 * 4 + 0) * 40 + kr] = (_Float16)bR[i].x;
      Bs[(nc4 * 4 + 1) * 40 + kr] = (_Float16)bR[i].y;
      Bs[(nc4 * 4 + 2) * 40 + kr] = (_Float16)bR[i].z;
      Bs[(nc4 * 4 + 3) * 40 + kr] = (_Float16)bR[i].w;
    }
  };

  v8f acc[4] = {};
  fetch(0);
  for (int kk = 0; kk < KD; kk += 32) {
    stage();
    __syncthreads();
    if (kk + 32 < KD) fetch(kk + 32);
    v16h af = load_a_frag(&As[wm * 16 * 40], 40);
#pragma unroll
    for (int j = 0; j < 4; ++j) {
      v16h bf = load_b_frag(&Bs[(wn + j) * 16 * 40], 40);
      acc[j] = wmma_f16(af, bf, acc[j]);
    }
    __syncthreads();
  }

#pragma unroll
  for (int j = 0; j < 4; ++j)
#pragma unroll
    for (int r = 0; r < 8; ++r)
      out[(size_t)(m0 + wm * 16 + r + 8 * half) * EMB + n0 + (wn + j) * 16 +
          mr] = acc[j][r];
}

// ============================================================================
extern "C" void kernel_launch(void* const* d_in, const int* in_sizes, int n_in,
                              void* d_out, int out_size, void* d_ws,
                              size_t ws_size, hipStream_t stream) {
  const float* xq   = (const float*)d_in[0];
  const int*   xpos = (const int*)d_in[1];
  const int*   seg  = (const int*)d_in[2];
  const float* wqkv = (const float*)d_in[3];
  const float* wout = (const float*)d_in[4];
  float* out = (float*)d_out;

  _Float16* qbuf = (_Float16*)d_ws;                                  // B*NH*S*D
  _Float16* kbuf = qbuf + (size_t)BATCH * NH * SEQ * HD;             // B*NKV*S*D
  _Float16* vbuf = kbuf + (size_t)BATCH * NKVH * SEQ * HD;           // B*NKV*S*D
  _Float16* abuf = vbuf + (size_t)BATCH * NKVH * SEQ * HD;           // B*S*NH*D

  qkv_rope_kernel<<<dim3(MROWS / 64, 32), 256, 0, stream>>>(xq, xpos, wqkv,
                                                            qbuf, kbuf, vbuf);
  attn_kernel<<<dim3(SEQ / 128, BATCH * NH), 256, 0, stream>>>(qbuf, kbuf, vbuf,
                                                               seg, abuf);
  outproj_kernel<<<dim3(MROWS / 64, EMB / 128), 256, 0, stream>>>(abuf, wout,
                                                                  out);
}
